// Blocks_86096914416140
// MI455X (gfx1250) — matrix operand order repaired and checked by reading, not compiled, not run on hardware
//
#include <hip/hip_runtime.h>
#include <hip/hip_bf16.h>

#define BB 32
#define CC 512
#define TT 32
#define NBL 32
#define NN (BB*TT)       // 1024 GEMM columns (b,t)
#define TLEN (NBL*TT)    // 1024 timesteps

typedef __attribute__((ext_vector_type(16))) __bf16       v16bf;
typedef __attribute__((ext_vector_type(8)))  float        v8f;
typedef __attribute__((ext_vector_type(4)))  unsigned int u32x4;

union V16 { u32x4 q[2]; v16bf v; };
union Acc { v8f v; float f[8]; };

// ---------------------------------------------------------------------------
// Split rec_weight into hi/lo bf16 (Markidis split): W ~= hi + lo, exact 0/1
// spikes make the GEMM accurate to ~16 mantissa bits with f32 accumulation.
// ---------------------------------------------------------------------------
__global__ void prep_w_kernel(const float* __restrict__ W,
                              __bf16* __restrict__ whi,
                              __bf16* __restrict__ wlo) {
  int i = blockIdx.x * blockDim.x + threadIdx.x;
  if (i >= CC * CC) return;
  float w = W[i];
  __bf16 h = (__bf16)w;
  whi[i] = h;
  wlo[i] = (__bf16)(w - (float)h);
}

// ---------------------------------------------------------------------------
// R[i][n] = sum_j (Whi[i][j] + Wlo[i][j]) * St[n][j],  n = b*32 + t.
// One wave per 16x16 tile; K-loop 512 in steps of 32; 2 WMMAs per step.
// A layout (16-bit 16x32): lane m=lane&15; kbase 0/8 for lane<16/>=16;
//   elems 0..7 -> K=kbase+i, elems 8..15 -> K=kbase+16+(i-8)  (two 16B loads).
// B layout (16-bit 32x16): lane n=lane&15; kbase 0/8; elems follow the same K order as A
//   (two 16B loads) -- St stored [n][j] to enable this.
// ---------------------------------------------------------------------------
__global__ __launch_bounds__(256) void gemm_rec_kernel(
    const __bf16* __restrict__ whi, const __bf16* __restrict__ wlo,
    const __bf16* __restrict__ st, float* __restrict__ R) {
  const int lane = threadIdx.x & 31;
  const int w    = blockIdx.x * 8 + (threadIdx.x >> 5);   // 2048 waves
  const int m0   = (w & 31) * 16;                         // 32 M tiles
  const int n0   = (w >> 5) * 16;                         // 64 N tiles
  const int kbA  = (lane < 16) ? 0 : 8;
  const int kbB  = (lane < 16) ? 0 : 8;
  const __bf16* wh = whi + (m0 + (lane & 15)) * CC;
  const __bf16* wl = wlo + (m0 + (lane & 15)) * CC;
  const __bf16* sb = st  + (n0 + (lane & 15)) * CC;

  v8f acc = {0.f, 0.f, 0.f, 0.f, 0.f, 0.f, 0.f, 0.f};
  for (int k = 0; k < CC; k += 32) {
    V16 ah, al, bm;
    ah.q[0] = *(const u32x4*)(wh + k + kbA);
    ah.q[1] = *(const u32x4*)(wh + k + kbA + 16);
    al.q[0] = *(const u32x4*)(wl + k + kbA);
    al.q[1] = *(const u32x4*)(wl + k + kbA + 16);
    bm.q[0] = *(const u32x4*)(sb + k + kbB);
    bm.q[1] = *(const u32x4*)(sb + k + kbB + 16);
    if (k + 32 < CC) {
      __builtin_prefetch(wh + k + 32, 0, 3);   // global_prefetch_b8
      __builtin_prefetch(sb + k + 32, 0, 3);
    }
    acc = __builtin_amdgcn_wmma_f32_16x16x32_bf16(false, ah.v, false, bm.v,
                                                  (short)0, acc, false, false);
    acc = __builtin_amdgcn_wmma_f32_16x16x32_bf16(false, al.v, false, bm.v,
                                                  (short)0, acc, false, false);
  }
  // C/D layout: VGPR r of lane l -> (M = r + 8*(l>>4), N = l&15)
  Acc a; a.v = acc;
  const int rbase = m0 + 8 * (lane >> 4);
  const int col   = n0 + (lane & 15);
#pragma unroll
  for (int r = 0; r < 8; ++r) R[(rbase + r) * NN + col] = a.f[r];
}

// ---------------------------------------------------------------------------
// One wave32 per (b,c), lane = t. Handles masking/adaptation state, the
// beta linear-recurrence scan (mem), the double prefix-sum (z), spike output,
// and writes spikes transposed in bf16 for the next block's GEMM B operand.
// ---------------------------------------------------------------------------
__global__ __launch_bounds__(256) void pw_block_kernel(
    const float* __restrict__ x, const float* __restrict__ beta_raw,
    const float* __restrict__ p_raw, const float* __restrict__ b_raw,
    const float* __restrict__ R, float* __restrict__ zbuf,
    float* __restrict__ mem_last, float* __restrict__ a_carry,
    __bf16* __restrict__ st, float* __restrict__ out, int nb, int first) {
  const int lane = threadIdx.x & 31;
  const int gw   = blockIdx.x * 8 + (threadIdx.x >> 5);   // 16384 waves
  const int b    = gw >> 9;
  const int c    = gw & 511;
  const int t    = lane;
  const int bc   = b * CC + c;

  const float beta = fminf(fmaxf(beta_raw[c], 0.001f), 0.999f);
  const float p    = fminf(fabsf(p_raw[c]), 0.999f);
  const float bth  = fminf(fmaxf(fabsf(b_raw[c]), 0.001f), 1.0f);
  const float ppow = powf(p, (float)(t + 1));             // p^(t+1)
  const float xv   = x[(long)bc * TLEN + nb * TT + t];

  float u, v_init, acur;
  if (!first) {
    const float zp  = zbuf[bc * TT + t];
    const float Rv  = R[c * NN + b * TT + t];
    const float spd = (zp == 1.0f) ? 1.0f : 0.0f;         // forward of _g(z)
    const float mask = __any(spd != 0.0f) ? 1.0f : 0.0f;  // mask_f
    u      = (zp < mask) ? 0.0f : (xv + Rv);
    v_init = mem_last[bc] * (1.0f - mask);
    const float acp = a_carry[bc];                        // a_kernel[t]=p^(t+1)*acp
    float s1 = spd * ppow * acp;                          // sum a_kernel*sp_det
    float s2 = (zp > 1.0f) ? 1.0f : 0.0f;                 // decay_steps
    for (int off = 16; off; off >>= 1) {
      s1 += __shfl_xor(s1, off, 32);
      s2 += __shfl_xor(s2, off, 32);
    }
    const float a_at_spike = s1 + 1.0f / p;
    const float new_a = a_at_spike * powf(p, s2);
    acur = powf(p, 32.0f) * acp * (1.0f - mask) + new_a * mask;
  } else {
    u = xv; v_init = 0.0f; acur = 0.0f;                   // block 0: v_th = 1
  }
  const float vth = 1.0f + bth * ppow * acur;

  // mem[t] = beta*mem[t-1] + u[t], mem[-1]=v_init: Kogge-Stone on (A,B) ops.
  float v = u, f = beta;
#pragma unroll
  for (int off = 1; off < 32; off <<= 1) {
    float pv = __shfl_up(v, off, 32);
    float pf = __shfl_up(f, off, 32);
    if (lane >= off) { v += f * pv; f *= pf; }            // f ends as beta^(t+1)
  }
  const float mem = v + f * v_init;
  const float fs  = (mem - vth > 0.0f) ? 1.0f : 0.0f;

  // z = double inclusive prefix-sum of fs  (Zm einsum)
  float S = fs;
#pragma unroll
  for (int off = 1; off < 32; off <<= 1) {
    float pv = __shfl_up(S, off, 32);
    if (lane >= off) S += pv;
  }
  float z = S;
#pragma unroll
  for (int off = 1; off < 32; off <<= 1) {
    float pv = __shfl_up(z, off, 32);
    if (lane >= off) z += pv;
  }
  const float spike = (z == 1.0f) ? 1.0f : 0.0f;

  out[(long)bc * TLEN + nb * TT + t] = spike;
  zbuf[bc * TT + t] = z;
  st[(b * TT + t) * CC + c] = (__bf16)spike;              // transposed for GEMM B
  if (lane == 31) mem_last[bc] = mem;
  if (lane == 0)  a_carry[bc] = acur;
}

// ---------------------------------------------------------------------------
extern "C" void kernel_launch(void* const* d_in, const int* in_sizes, int n_in,
                              void* d_out, int out_size, void* d_ws, size_t ws_size,
                              hipStream_t stream) {
  const float* x        = (const float*)d_in[0];   // (32,512,1024)
  const float* beta_raw = (const float*)d_in[1];   // (512,)
  const float* W        = (const float*)d_in[2];   // (512,512)
  const float* p_raw    = (const float*)d_in[3];   // (512,)
  const float* b_raw    = (const float*)d_in[4];   // (512,)
  float* out = (float*)d_out;

  char* ws = (char*)d_ws;
  __bf16* whi  = (__bf16*)(ws);                          // 512 KB
  __bf16* wlo  = (__bf16*)(ws + (512u << 10));           // 512 KB
  __bf16* st   = (__bf16*)(ws + (1u << 20));             // 1 MB  [n][c] spikes
  float*  R    = (float*) (ws + (2u << 20));             // 2 MB  [c][n]
  float*  zbuf = (float*) (ws + (4u << 20));             // 2 MB  [b,c,t]
  float*  memL = (float*) (ws + (6u << 20));             // 64 KB [b,c]
  float*  aC   = (float*) (ws + (6u << 20) + (64u << 10)); // 64 KB [b,c]

  prep_w_kernel<<<(CC * CC + 255) / 256, 256, 0, stream>>>(W, whi, wlo);

  // Block 0: no recurrence, v_th = 1, zero initial state.
  pw_block_kernel<<<2048, 256, 0, stream>>>(x, beta_raw, p_raw, b_raw,
                                            R, zbuf, memL, aC, st, out, 0, 1);
  for (int nb = 1; nb < NBL; ++nb) {
    gemm_rec_kernel<<<256, 256, 0, stream>>>(whi, wlo, st, R);
    pw_block_kernel<<<2048, 256, 0, stream>>>(x, beta_raw, p_raw, b_raw,
                                              R, zbuf, memL, aC, st, out, nb, 0);
  }
}
